// FramewiseAdaptiveGammatoneFB_49701361549326
// MI455X (gfx1250) — matrix operand order repaired and checked by reading, not compiled, not run on hardware
//
#include <hip/hip_runtime.h>
#include <math.h>

typedef _Float16 h16_t;
typedef __attribute__((ext_vector_type(16))) _Float16 v16h;
typedef __attribute__((ext_vector_type(8)))  _Float16 v8h;
typedef __attribute__((ext_vector_type(8)))  float    v8f;

#define TSTEPS 19
#define FS     16000
#define NFFT   1024
#define WINLEN 842
#define FBINS  513
#define FPAD   528     // FBINS padded to multiple of 16
#define NB     100
#define HID    128
#define NG     384     // 3*HID
#define KX     224     // 2*NB = 200 padded to multiple of 32
#define W3R    112     // NB padded to multiple of 16

// ---------------- WMMA fragment loaders (CDNA5 16-bit layouts) ----------------
// A: 16x32 (MxK) row-major, lanes 0-15: M=lane, K base 0; lanes 16-31: M=lane-16, K base 8.
// Lane halves 0..7 = K[base..base+7], halves 8..15 = K[base+16..base+23].
__device__ inline v16h load_a_frag(const h16_t* A, int lda, int ml, int hs, int k0) {
    const h16_t* p = A + (size_t)ml * lda + k0 + hs * 8;
    v8h a0 = *(const v8h*)p;
    v8h a1 = *(const v8h*)(p + 16);
    v16h a;
#pragma unroll
    for (int i = 0; i < 8; i++) { a[i] = a0[i]; a[i + 8] = a1[i]; }
    return a;
}
// B: 32x16 (KxN), stored N-major (Bt[n][k], row stride ldb). Lane n = lane&15,
// K base = (lane>>4)*8; lane's halves 0..7 = Bt[n][base..base+7], 8..15 = Bt[n][base+16..base+23].
__device__ inline v16h load_b_frag(const h16_t* Bt, int ldb, int nl, int hs, int k0) {
    const h16_t* p = Bt + (size_t)nl * ldb + k0 + hs * 8;
    v8h b0 = *(const v8h*)p;
    v8h b1 = *(const v8h*)(p + 16);
    v16h b;
#pragma unroll
    for (int i = 0; i < 8; i++) { b[i] = b0[i]; b[i + 8] = b1[i]; }
    return b;
}

// Async memory->LDS staging of a contiguous region (all threads of the block
// participate; caller must s_wait_asynccnt + barrier afterwards).
// Uses CDNA5 GLOBAL_LOAD_ASYNC_TO_LDS_B128 (tracked by ASYNCcnt).
__device__ inline void async_stage(const void* gsrc, void* lds_dst, int bytes) {
    unsigned long long gbase = (unsigned long long)gsrc;
    unsigned lbase = (unsigned)(unsigned long long)lds_dst;   // low 32 bits = LDS offset
    for (int off = threadIdx.x * 16; off < bytes; off += blockDim.x * 16) {
        unsigned long long ga = gbase + (unsigned)off;
        unsigned la = lbase + (unsigned)off;
        asm volatile("global_load_async_to_lds_b128 %0, %1, off"
                     :: "v"(la), "v"(ga) : "memory");
    }
}
__device__ inline void async_wait_barrier() {
    asm volatile("s_wait_asynccnt 0x0" ::: "memory");
    __syncthreads();
}

// ---------------- prep kernels ----------------
__global__ void basis_kernel(h16_t* cosT, h16_t* sinT) {
    int idx = blockIdx.x * blockDim.x + threadIdx.x;
    if (idx >= FPAD * NFFT) return;
    int f = idx / NFFT, k = idx % NFFT;
    float ang = 6.28318530717958647692f * (float)((k * f) % NFFT) / (float)NFFT;
    float s, c;
    __sincosf(ang, &s, &c);
    cosT[idx] = (h16_t)c;          // Re basis
    sinT[idx] = (h16_t)(-s);       // Im basis (e^{-i theta})
}

__device__ inline float erb_rate_f(float f) {
    return 21.4f * log10f(4.37f * f / 1000.0f + 1.0f);
}

__global__ void band_kernel(float* bands) {  // bands: [0..127]=fc [128..255]=q0 [256..383]=dq
    int n = threadIdx.x;
    if (n >= NB) return;
    const float fmin = 50.0f, fmax = 0.9f * (float)FS * 0.5f;  // 7200
    float E0 = erb_rate_f(fmin), E1 = erb_rate_f(fmax);
    float E  = E0 + (E1 - E0) * (float)n / (float)(NB - 1);
    float fc = (powf(10.0f, E / 21.4f) - 1.0f) * 1000.0f / 4.37f;
    float erbhz = 24.7f * (4.37f * fc / 1000.0f + 1.0f);
    float q0 = fc / (1.019f * erbhz);
    float En = (E - E0) / (E1 - E0 + 1e-12f);
    float dq = fmaxf(2.0f * (0.5f + 0.5f * En), 0.001f);
    bands[n] = fc; bands[128 + n] = q0; bands[256 + n] = dq;
}

// f32 -> f16 with row/col zero-padding (dst is drows x dcols, src srows x scols)
__global__ void convw_kernel(const float* src, h16_t* dst, int drows, int srows,
                             int scols, int dcols) {
    int idx = blockIdx.x * blockDim.x + threadIdx.x;
    if (idx >= drows * dcols) return;
    int r = idx / dcols, c = idx % dcols;
    float v = (r < srows && c < scols) ? src[(size_t)r * scols + c] : 0.0f;
    dst[idx] = (h16_t)v;
}

__global__ void init_kernel(h16_t* frames, h16_t* xbuf, float* h32, h16_t* h16b,
                            float* qprev, const float* bands, int B) {
    int idx = blockIdx.x * blockDim.x + threadIdx.x;
    int nf = B * NFFT;
    if (idx < nf) { frames[idx] = (h16_t)0.0f; return; }
    idx -= nf; nf = B * KX;
    if (idx < nf) { xbuf[idx] = (h16_t)0.0f; return; }
    idx -= nf; nf = B * HID;
    if (idx < nf) { h32[idx] = 0.0f; h16b[idx] = (h16_t)0.0f; return; }
    idx -= nf; nf = B * NB;
    if (idx < nf) { qprev[idx] = bands[128 + (idx % NB)]; }
}

// ---------------- per-step kernels ----------------
__global__ void frame_kernel(const float* wav, h16_t* frames, int t, int B) {
    int idx = blockIdx.x * blockDim.x + threadIdx.x;
    if (idx >= B * WINLEN) return;
    int b = idx / WINLEN, k = idx % WINLEN;
    float w = 0.5f * (1.0f - __cosf(6.28318530717958647692f * (float)k / (float)WINLEN));
    frames[(size_t)b * NFFT + k] = (h16_t)(wav[(size_t)b * FS + t * WINLEN + k] * w);
}

// DFT via WMMA: (B x 1024) fr16 @ (1024 x FPAD) {cos,sin}; write X (complex) + |X|.
// A tile (16 x 1024, 32KB) is async-staged to LDS once per block and shared by
// the 4 waves (4 freq tiles), cutting A global traffic 4x.
__global__ void dft_kernel(const h16_t* frames, const h16_t* cosT, const h16_t* sinT,
                           float* xmag, float* outX, int t, int B) {
    __shared__ h16_t sA[16 * NFFT];
    int wave = threadIdx.x >> 5, lane = threadIdx.x & 31;
    int ftile = blockIdx.x * 4 + wave;
    int mtile = blockIdx.y;
    int ml = lane & 15, hs = lane >> 4;

    async_stage(frames + (size_t)(mtile * 16) * NFFT, sA, 16 * NFFT * 2);
    async_wait_barrier();

    if (ftile < FPAD / 16) {                         // wave-uniform guard
        const h16_t* Cb = cosT + (size_t)(ftile * 16) * NFFT;
        const h16_t* Sb = sinT + (size_t)(ftile * 16) * NFFT;
        v8f ar = {}, ai = {};
        for (int k0 = 0; k0 < NFFT; k0 += 32) {
            __builtin_prefetch((const void*)(Cb + (size_t)ml * NFFT + k0 + 128), 0, 1);
            v16h a  = load_a_frag(sA, NFFT, ml, hs, k0);   // ds_load from LDS
            v16h bc = load_b_frag(Cb, NFFT, ml, hs, k0);
            ar = __builtin_amdgcn_wmma_f32_16x16x32_f16(false, a, false, bc, (short)0, ar, false, false);
            v16h bs = load_b_frag(Sb, NFFT, ml, hs, k0);
            ai = __builtin_amdgcn_wmma_f32_16x16x32_f16(false, a, false, bs, (short)0, ai, false, false);
        }
        int f = ftile * 16 + ml;
#pragma unroll
        for (int i = 0; i < 8; i++) {
            int b = mtile * 16 + hs * 8 + i;
            float re = ar[i], im = ai[i];
            if (f < FBINS) {
                size_t xi = ((size_t)(b * TSTEPS + t) * FBINS + f) * 2;
                outX[xi] = re; outX[xi + 1] = im;
                xmag[(size_t)b * FPAD + f] = sqrtf(re * re + im * im);
            }
        }
    }
}

// adaptive Gaussian filterbank: per (b,band) reduction over 513 bins
__global__ void fb_kernel(const float* xmag, const float* qprev, const float* bands,
                          h16_t* xbuf, float* outY, float* outQ, int t, int B) {
    int idx = blockIdx.x * blockDim.x + threadIdx.x;
    if (idx >= B * NB) return;
    int b = idx / NB, n = idx % NB;
    float fc = bands[n];
    float q  = qprev[(size_t)b * NB + n];
    float bw = fc / (q + 1e-8f) + 1e-8f;
    float inv = 1.0f / bw;
    const float* mg = xmag + (size_t)b * FPAD;
    float s = 0.0f, ws = 0.0f;
    for (int f = 0; f < FBINS; f++) {
        float d = ((float)f * 15.625f - fc) * inv;   // FS/2/(FBINS-1) = 15.625 Hz/bin
        float w = __expf(-0.5f * d * d);
        ws += w;
        s  += w * mg[f];
    }
    float Y = s / (ws + 1e-8f);
    if (!(Y > -1e30f && Y < 1e30f)) Y = 0.0f;        // nan/inf -> 0
    float yc = log1pf(fmaxf(Y, 0.0f));
    outY[(size_t)(b * TSTEPS + t) * NB + n] = Y;
    outQ[(size_t)(b * TSTEPS + t) * NB + n] = q;     // Q_prev is collected
    xbuf[(size_t)b * KX + n]      = (h16_t)yc;
    xbuf[(size_t)b * KX + NB + n] = (h16_t)(0.2f * yc);
}

// generic (B x K) @ Bt(N x K)^T + bias -> C (f32), one 16x16 tile per wave.
// Requires lda == K (contiguous A tile); A tile async-staged to LDS per block.
__global__ void gemm_bias_kernel(const h16_t* A, const h16_t* Bt, int ldb,
                                 const float* bias, float* C, int ldc, int K, int Ntiles) {
    __shared__ h16_t sA[16 * KX];                    // max K = KX = 224
    int wave = threadIdx.x >> 5, lane = threadIdx.x & 31;
    int ntile = blockIdx.x * 4 + wave;
    int mtile = blockIdx.y;
    int ml = lane & 15, hs = lane >> 4;

    async_stage(A + (size_t)(mtile * 16) * K, sA, 16 * K * 2);
    async_wait_barrier();

    if (ntile < Ntiles) {                            // wave-uniform guard
        const h16_t* Bb = Bt + (size_t)(ntile * 16) * ldb;
        v8f acc = {};
        for (int k0 = 0; k0 < K; k0 += 32) {
            v16h a = load_a_frag(sA, K, ml, hs, k0);
            v16h b = load_b_frag(Bb, ldb, ml, hs, k0);
            acc = __builtin_amdgcn_wmma_f32_16x16x32_f16(false, a, false, b, (short)0, acc, false, false);
        }
        float bv = bias[ntile * 16 + ml];
        float* crow = C + (size_t)(mtile * 16 + hs * 8) * ldc + ntile * 16 + ml;
#pragma unroll
        for (int i = 0; i < 8; i++) crow[(size_t)i * ldc] = acc[i] + bv;
    }
}

// GRU gate math (r,z,n torch order), updates h (f32 + f16 copy)
__global__ void gru_gate_kernel(const float* gi, const float* gh, float* h32,
                                h16_t* h16b, int B) {
    int idx = blockIdx.x * blockDim.x + threadIdx.x;
    if (idx >= B * HID) return;
    int b = idx / HID, j = idx % HID;
    const float* gib = gi + (size_t)b * NG;
    const float* ghb = gh + (size_t)b * NG;
    float r = 1.0f / (1.0f + __expf(-(gib[j]           + ghb[j])));
    float z = 1.0f / (1.0f + __expf(-(gib[HID + j]     + ghb[HID + j])));
    float n = tanhf(gib[2 * HID + j] + r * ghb[2 * HID + j]);
    float hn = (1.0f - z) * n + z * h32[idx];
    h32[idx] = hn;
    h16b[idx] = (h16_t)hn;
}

// LayerNorm(128) + SiLU; one wave per row, shfl_xor reduction (wave32)
__global__ void ln_silu_kernel(const float* Z, const float* g, const float* be,
                               h16_t* out, int B) {
    int wave = threadIdx.x >> 5, lane = threadIdx.x & 31;
    int row = blockIdx.x * (blockDim.x >> 5) + wave;
    if (row >= B) return;
    const float* zr = Z + (size_t)row * HID;
    float v[4];
    float s = 0.0f;
#pragma unroll
    for (int i = 0; i < 4; i++) { v[i] = zr[lane + 32 * i]; s += v[i]; }
#pragma unroll
    for (int o = 16; o > 0; o >>= 1) s += __shfl_xor(s, o, 32);
    float mu = s * (1.0f / (float)HID);
    float qv = 0.0f;
#pragma unroll
    for (int i = 0; i < 4; i++) { float d = v[i] - mu; qv += d * d; }
#pragma unroll
    for (int o = 16; o > 0; o >>= 1) qv += __shfl_xor(qv, o, 32);
    float inv = rsqrtf(qv * (1.0f / (float)HID) + 1e-5f);
#pragma unroll
    for (int i = 0; i < 4; i++) {
        int j = lane + 32 * i;
        float y = (v[i] - mu) * inv * g[j] + be[j];
        float si = y / (1.0f + __expf(-y));
        out[(size_t)row * HID + j] = (h16_t)si;
    }
}

// W3 head: (B x 128) @ (112 x 128)^T, tanh, Q = clip(q0 + dq*delta)
__global__ void qhead_kernel(const h16_t* A, const h16_t* Bt, const float* b3,
                             const float* bands, float* qprev, int B) {
    __shared__ h16_t sA[16 * HID];
    int wave = threadIdx.x >> 5, lane = threadIdx.x & 31;
    int ntile = blockIdx.x * 4 + wave;
    int mtile = blockIdx.y;
    int ml = lane & 15, hs = lane >> 4;

    async_stage(A + (size_t)(mtile * 16) * HID, sA, 16 * HID * 2);
    async_wait_barrier();

    if (ntile < W3R / 16) {                          // wave-uniform guard
        const h16_t* Bb = Bt + (size_t)(ntile * 16) * HID;
        v8f acc = {};
        for (int k0 = 0; k0 < HID; k0 += 32) {
            v16h a = load_a_frag(sA, HID, ml, hs, k0);
            v16h b = load_b_frag(Bb, HID, ml, hs, k0);
            acc = __builtin_amdgcn_wmma_f32_16x16x32_f16(false, a, false, b, (short)0, acc, false, false);
        }
        int n = ntile * 16 + ml;
        if (n < NB) {
            float q0 = bands[128 + n], dq = bands[256 + n], bv = b3[n];
#pragma unroll
            for (int i = 0; i < 8; i++) {
                int b = mtile * 16 + hs * 8 + i;
                float delta = tanhf(acc[i] + bv);
                float Q = fminf(fmaxf(q0 + dq * delta, 0.05f), 30.0f);
                qprev[(size_t)b * NB + n] = Q;
            }
        }
    }
}

// ---------------- host launcher ----------------
extern "C" void kernel_launch(void* const* d_in, const int* in_sizes, int n_in,
                              void* d_out, int out_size, void* d_ws, size_t ws_size,
                              hipStream_t stream) {
    (void)n_in; (void)out_size;
    const float* wav  = (const float*)d_in[0];
    const float* W_ih = (const float*)d_in[1];
    const float* W_hh = (const float*)d_in[2];
    const float* b_ih = (const float*)d_in[3];
    const float* b_hh = (const float*)d_in[4];
    const float* W1   = (const float*)d_in[5];
    const float* b1   = (const float*)d_in[6];
    const float* g1   = (const float*)d_in[7];
    const float* be1  = (const float*)d_in[8];
    const float* W2   = (const float*)d_in[9];
    const float* b2   = (const float*)d_in[10];
    const float* g2   = (const float*)d_in[11];
    const float* be2  = (const float*)d_in[12];
    const float* W3   = (const float*)d_in[13];
    const float* b3   = (const float*)d_in[14];

    const int B = in_sizes[0] / FS;                  // 512 in reference
    float* outY = (float*)d_out;
    float* outQ = outY + (size_t)B * TSTEPS * NB;
    float* outX = outQ + (size_t)B * TSTEPS * NB;    // complex64 interleaved

    // workspace carve-out (256B aligned)
    size_t off = 0;
    auto take = [&](size_t bytes) { size_t o = off; off = (off + bytes + 255) & ~(size_t)255; return o; };
    char* ws = (char*)d_ws;
    h16_t* cosT   = (h16_t*)(ws + take((size_t)FPAD * NFFT * 2));
    h16_t* sinT   = (h16_t*)(ws + take((size_t)FPAD * NFFT * 2));
    h16_t* frames = (h16_t*)(ws + take((size_t)B * NFFT * 2));
    float* xmag   = (float*)(ws + take((size_t)B * FPAD * 4));
    h16_t* xbuf   = (h16_t*)(ws + take((size_t)B * KX * 2));
    float* h32    = (float*)(ws + take((size_t)B * HID * 4));
    h16_t* h16b   = (h16_t*)(ws + take((size_t)B * HID * 2));
    float* gi     = (float*)(ws + take((size_t)B * NG * 4));
    float* gh     = (float*)(ws + take((size_t)B * NG * 4));
    float* zbuf   = (float*)(ws + take((size_t)B * HID * 4));
    h16_t* a16    = (h16_t*)(ws + take((size_t)B * HID * 2));
    h16_t* Wih16  = (h16_t*)(ws + take((size_t)NG * KX * 2));
    h16_t* Whh16  = (h16_t*)(ws + take((size_t)NG * HID * 2));
    h16_t* W116   = (h16_t*)(ws + take((size_t)HID * HID * 2));
    h16_t* W216   = (h16_t*)(ws + take((size_t)HID * HID * 2));
    h16_t* W316   = (h16_t*)(ws + take((size_t)W3R * HID * 2));
    float* qprev  = (float*)(ws + take((size_t)B * NB * 4));
    float* bands  = (float*)(ws + take((size_t)384 * 4));
    if (off > ws_size) return;                       // workspace too small: bail safely

    const int Mt = B / 16;                           // batch tiles
    dim3 blk256(256), blk128(128);

    // ---- prep ----
    basis_kernel<<<(FPAD * NFFT + 255) / 256, blk256, 0, stream>>>(cosT, sinT);
    band_kernel<<<1, 128, 0, stream>>>(bands);
    convw_kernel<<<(NG * KX   + 255) / 256, blk256, 0, stream>>>(W_ih, Wih16, NG, NG, 2 * NB, KX);
    convw_kernel<<<(NG * HID  + 255) / 256, blk256, 0, stream>>>(W_hh, Whh16, NG, NG, HID, HID);
    convw_kernel<<<(HID * HID + 255) / 256, blk256, 0, stream>>>(W1, W116, HID, HID, HID, HID);
    convw_kernel<<<(HID * HID + 255) / 256, blk256, 0, stream>>>(W2, W216, HID, HID, HID, HID);
    convw_kernel<<<(W3R * HID + 255) / 256, blk256, 0, stream>>>(W3, W316, W3R, NB, HID, HID);
    {
        int n = B * (NFFT + KX + HID + NB);
        init_kernel<<<(n + 255) / 256, blk256, 0, stream>>>(frames, xbuf, h32, h16b, qprev, bands, B);
    }

    // ---- sequential scan over timesteps ----
    for (int t = 0; t < TSTEPS; t++) {
        frame_kernel<<<(B * WINLEN + 255) / 256, blk256, 0, stream>>>(wav, frames, t, B);
        dft_kernel<<<dim3((FPAD / 16 + 3) / 4, Mt), blk128, 0, stream>>>(
            frames, cosT, sinT, xmag, outX, t, B);
        fb_kernel<<<(B * NB + 255) / 256, blk256, 0, stream>>>(
            xmag, qprev, bands, xbuf, outY, outQ, t, B);
        // GRU: gi = x @ W_ih^T + b_ih ; gh = h @ W_hh^T + b_hh
        gemm_bias_kernel<<<dim3((NG / 16 + 3) / 4, Mt), blk128, 0, stream>>>(
            xbuf, Wih16, KX, b_ih, gi, NG, KX, NG / 16);
        gemm_bias_kernel<<<dim3((NG / 16 + 3) / 4, Mt), blk128, 0, stream>>>(
            h16b, Whh16, HID, b_hh, gh, NG, HID, NG / 16);
        gru_gate_kernel<<<(B * HID + 255) / 256, blk256, 0, stream>>>(gi, gh, h32, h16b, B);
        // MLP head: Linear -> LN -> SiLU (x2) -> Linear -> tanh -> Q update
        gemm_bias_kernel<<<dim3((HID / 16 + 3) / 4, Mt), blk128, 0, stream>>>(
            h16b, W116, HID, b1, zbuf, HID, HID, HID / 16);
        ln_silu_kernel<<<(B + 7) / 8, blk256, 0, stream>>>(zbuf, g1, be1, a16, B);
        gemm_bias_kernel<<<dim3((HID / 16 + 3) / 4, Mt), blk128, 0, stream>>>(
            a16, W216, HID, b2, zbuf, HID, HID, HID / 16);
        ln_silu_kernel<<<(B + 7) / 8, blk256, 0, stream>>>(zbuf, g2, be2, a16, B);
        qhead_kernel<<<dim3((W3R / 16 + 3) / 4, Mt), blk128, 0, stream>>>(
            a16, W316, b3, bands, qprev, B);
    }
}
